// QLoRALinear_3006477107228
// MI455X (gfx1250) — matrix operand order repaired and checked by reading, not compiled, not run on hardware
//
#include <hip/hip_runtime.h>
#include <hip/hip_bf16.h>

// ---------------------------------------------------------------------------
// QLoRA linear for MI455X (gfx1250, wave32, WMMA).
//   out = x @ W_eff^T,  W_eff = NF4dequant(codes,absmax) + 0.25 * (B @ A)
// Pipeline:
//   k1: x f32 -> f16                       (ws[0 .. M*K))
//   k2: dequant + LoRA fold -> W_eff f16   (ws[M*K .. M*K+N*K))
//   k3: tiled f16 WMMA GEMM (f32 accum), 128x128 tiles, double-buffered LDS
//       fed by GLOBAL_LOAD_ASYNC_TO_LDS_B128 (ASYNCcnt path).
// x_f16 (64MB) + W_f16 (32MB) both resident in 192MB L2 -> GEMM re-reads are
// L2 hits; HBM traffic ~ one pass over inputs + outputs (~320MB, ~14us),
// compute-bound on the v_wmma_f32_16x16x32_f16 pipe.
// K and N are compile-time constants so all tile/epilogue addressing folds
// into immediate offsets (no 64-bit VALU address chains).
// ---------------------------------------------------------------------------

typedef __attribute__((ext_vector_type(16))) _Float16 v16h;
typedef __attribute__((ext_vector_type(8)))  _Float16 v8h;
typedef __attribute__((ext_vector_type(4)))  _Float16 v4h;
typedef __attribute__((ext_vector_type(8)))  float    v8f;

__constant__ float NF4C[16] = {
    -1.0f, -0.6961928009986877f, -0.5250730514526367f, -0.39491748809814453f,
    -0.28444138169288635f, -0.18477343022823334f, -0.09105003625154495f, 0.0f,
    0.07958029955625534f, 0.16093020141124725f, 0.24611230194568634f,
    0.33791524171829224f, 0.44070982933044434f, 0.5626170039176941f,
    0.7229568362236023f, 1.0f};

#define KC 4096          // D_in  (compile-time)
#define NC 4096          // D_out (compile-time)
#define RC 64            // LoRA rank
#define LDA 40           // LDS row stride in halves: 80B, 16B-aligned, bank-conflict-free
#define SCALING 0.25f    // lora_alpha / r = 16/64

#if __has_builtin(__builtin_amdgcn_global_load_async_to_lds_b128) && \
    __has_builtin(__builtin_amdgcn_s_wait_asynccnt)
#define HAVE_ASYNC_LDS 1
typedef int async_v4i __attribute__((__vector_size__(16)));
typedef __attribute__((address_space(1))) async_v4i* g_v4i_p;
typedef __attribute__((address_space(3))) async_v4i* l_v4i_p;
#define ASYNC_B128(g, l)                                             \
    __builtin_amdgcn_global_load_async_to_lds_b128(                  \
        (g_v4i_p)(g), (l_v4i_p)(l), 0, 0)
#else
#define HAVE_ASYNC_LDS 0
#endif

// ---- kernel 1: f32 -> f16 convert (4 elements per thread) -----------------
__global__ __launch_bounds__(256) void cvt_f32_to_f16(const float* __restrict__ x,
                                                      _Float16* __restrict__ y,
                                                      int n) {
    int i = (blockIdx.x * 256 + threadIdx.x) * 4;
    if (i + 3 >= n) return;
    float4 v = *(const float4*)(x + i);
    v4h h;
    h[0] = (_Float16)v.x; h[1] = (_Float16)v.y;
    h[2] = (_Float16)v.z; h[3] = (_Float16)v.w;
    *(v4h*)(y + i) = h;
}

// ---- kernel 2: NF4 dequant + LoRA fold ------------------------------------
// W_eff[n][k] = NF4[codes[n][k]] * absmax[n][k/64] + SCALING * sum_r B[n][r]*A[r][k]
__global__ __launch_bounds__(256) void dequant_fold(const int*   __restrict__ codes,
                                                    const float* __restrict__ absmax,
                                                    const float* __restrict__ lA,  // [r][K]
                                                    const float* __restrict__ lB,  // [N][r]
                                                    _Float16*    __restrict__ Wh)  // [N][K]
{
    const int k = blockIdx.x * 256 + threadIdx.x;
    const int n = blockIdx.y;
    const int idx = n * KC + k;            // < 2^24, 32-bit safe
    float w = NF4C[codes[idx] & 15] * absmax[n * (KC >> 6) + (k >> 6)];
    const float* Ak = lA + k;              // column k of A, stride KC
    const float* Bn = lB + n * RC;         // row n of B (uniform per block -> scalar loads)
    float acc = 0.0f;
#pragma unroll 8
    for (int rr = 0; rr < RC; ++rr)
        acc = fmaf(Bn[rr], Ak[rr * KC], acc);
    Wh[idx] = (_Float16)(w + SCALING * acc);
}

// ---- kernel 3: WMMA GEMM ---------------------------------------------------
// Out[M][NC] = Xh[M][KC] * Wh[NC][KC]^T  (both operands K-major)
// Block: 256 threads (8 waves), tile 128(M) x 128(N), K-step 32.
// Wave grid 2x4: each wave -> 64(M) x 32(N) = 4x2 WMMA accumulators.
__global__ __launch_bounds__(256) void qlora_gemm(const _Float16* __restrict__ Xh,
                                                  const _Float16* __restrict__ Wh,
                                                  float* __restrict__ Out) {
    __shared__ _Float16 lds[2 * 2 * 128 * LDA];   // [buf][A/B][128][LDA] = 40KB

    const int tid   = threadIdx.x;
    const int lane  = tid & 31;
    const int wv    = tid >> 5;
    const int wr    = wv >> 2;          // wave row   (0..1) -> 64 M-rows
    const int wc    = wv & 3;           // wave col   (0..3) -> 32 N-cols
    const int mlane = lane & 15;
    const int hsel  = lane >> 4;        // half-wave select
    const int tileM = blockIdx.y * 128;
    const int tileN = blockIdx.x * 128;
    const int nk    = KC >> 5;          // K/32 steps

    // global->LDS loader coords: 128 rows x 4 chunks(8 halves) = 512 chunks, 2/thread
    const int c0 = tid, c1 = tid + 256;
    const int r0 = c0 >> 2, cc0 = (c0 & 3) * 8;
    const int r1 = c1 >> 2, cc1 = (c1 & 3) * 8;
    const _Float16* xr0 = Xh + (size_t)(tileM + r0) * KC + cc0;
    const _Float16* xr1 = Xh + (size_t)(tileM + r1) * KC + cc1;
    const _Float16* wr0 = Wh + (size_t)(tileN + r0) * KC + cc0;
    const _Float16* wr1 = Wh + (size_t)(tileN + r1) * KC + cc1;
    const int la0 = r0 * LDA + cc0;     // LDS half-offsets within a tile
    const int la1 = r1 * LDA + cc1;

    v8f acc[4][2];
#pragma unroll
    for (int i = 0; i < 4; ++i)
#pragma unroll
        for (int j = 0; j < 2; ++j) acc[i][j] = (v8f)(0.0f);

    // prime buffer 0
    {
        _Float16* As = lds;
        _Float16* Bs = lds + 128 * LDA;
#if HAVE_ASYNC_LDS
        ASYNC_B128(xr0, As + la0);
        ASYNC_B128(xr1, As + la1);
        ASYNC_B128(wr0, Bs + la0);
        ASYNC_B128(wr1, Bs + la1);
        __builtin_amdgcn_s_wait_asynccnt(0);
#else
        *(uint4*)(As + la0) = *(const uint4*)(xr0);
        *(uint4*)(As + la1) = *(const uint4*)(xr1);
        *(uint4*)(Bs + la0) = *(const uint4*)(wr0);
        *(uint4*)(Bs + la1) = *(const uint4*)(wr1);
#endif
    }
    __syncthreads();

    for (int kt = 0; kt < nk; ++kt) {
        const int buf = kt & 1;
        _Float16* As = lds + buf * (2 * 128 * LDA);
        _Float16* Bs = As + 128 * LDA;
        const bool pre = (kt + 1 < nk);

#if HAVE_ASYNC_LDS
        // kick off next K-panel copies into the other buffer (overlaps WMMA)
        if (pre) {
            _Float16* As2 = lds + (buf ^ 1) * (2 * 128 * LDA);
            _Float16* Bs2 = As2 + 128 * LDA;
            const int ko = (kt + 1) * 32;
            ASYNC_B128(xr0 + ko, As2 + la0);
            ASYNC_B128(xr1 + ko, As2 + la1);
            ASYNC_B128(wr0 + ko, Bs2 + la0);
            ASYNC_B128(wr1 + ko, Bs2 + la1);
        }
#else
        uint4 pa0, pa1, pb0, pb1;
        if (pre) {
            const int ko = (kt + 1) * 32;
            pa0 = *(const uint4*)(xr0 + ko);
            pa1 = *(const uint4*)(xr1 + ko);
            pb0 = *(const uint4*)(wr0 + ko);
            pb1 = *(const uint4*)(wr1 + ko);
        }
#endif
        if (kt + 2 < nk) {   // L2 prefetch hint for k+2 panel
            const int ko2 = (kt + 2) * 32;
            __builtin_prefetch(xr0 + ko2, 0, 1);
            __builtin_prefetch(wr0 + ko2, 0, 1);
        }

        // A fragment (16x32 f16): lane m=mlane, halves = K{b..b+7, b+16..b+23}, b=8*hsel
        union Frag { v16h v; v8h h[2]; };
        Frag af[4], bf[2];
#pragma unroll
        for (int i = 0; i < 4; ++i) {
            const _Float16* p = As + (wr * 64 + i * 16 + mlane) * LDA + hsel * 8;
            af[i].h[0] = *(const v8h*)(p);
            af[i].h[1] = *(const v8h*)(p + 16);
        }
        // B fragment (32x16 f16): lane n=mlane, halves = K{b..b+7, b+16..b+23}, b=8*hsel of row n of W
#pragma unroll
        for (int j = 0; j < 2; ++j) {
            const _Float16* p = Bs + (wc * 32 + j * 16 + mlane) * LDA + hsel * 8;
            bf[j].h[0] = *(const v8h*)(p);
            bf[j].h[1] = *(const v8h*)(p + 16);
        }

#pragma unroll
        for (int i = 0; i < 4; ++i)
#pragma unroll
            for (int j = 0; j < 2; ++j)
                acc[i][j] = __builtin_amdgcn_wmma_f32_16x16x32_f16(
                    false, af[i].v, false, bf[j].v, (short)0, acc[i][j], false, false);

#if HAVE_ASYNC_LDS
        if (pre) __builtin_amdgcn_s_wait_asynccnt(0);
#else
        if (pre) {
            _Float16* As2 = lds + (buf ^ 1) * (2 * 128 * LDA);
            _Float16* Bs2 = As2 + 128 * LDA;
            *(uint4*)(As2 + la0) = pa0;
            *(uint4*)(As2 + la1) = pa1;
            *(uint4*)(Bs2 + la0) = pb0;
            *(uint4*)(Bs2 + la1) = pb1;
        }
#endif
        __syncthreads();
    }

    // Epilogue. D layout: m = vgpr + 8*hsel, n = mlane.
    // One base address; every store offset is a compile-time immediate.
    float* outp = Out + (size_t)(tileM + wr * 64 + hsel * 8) * NC
                      + (tileN + wc * 32 + mlane);
#pragma unroll
    for (int i = 0; i < 4; ++i)
#pragma unroll
        for (int j = 0; j < 2; ++j)
#pragma unroll
            for (int v = 0; v < 8; ++v)
                outp[(i * 16 + v) * NC + j * 16] = acc[i][j][v];
}

// ---------------------------------------------------------------------------
extern "C" void kernel_launch(void* const* d_in, const int* in_sizes, int n_in,
                              void* d_out, int out_size, void* d_ws, size_t ws_size,
                              hipStream_t stream) {
    const float* x      = (const float*)d_in[0];
    const int*   codes  = (const int*)d_in[1];
    const float* absmax = (const float*)d_in[2];
    const float* lora_A = (const float*)d_in[3];
    const float* lora_B = (const float*)d_in[4];
    float*       out    = (float*)d_out;

    const int M = in_sizes[0] / KC;      // B*S = 8192

    _Float16* xh = (_Float16*)d_ws;                  // [M][KC] f16 (64 MB)
    _Float16* wh = xh + (size_t)M * KC;              // [NC][KC] f16 (32 MB)

    // k1: x -> f16
    cvt_f32_to_f16<<<(M * KC) / 1024, 256, 0, stream>>>(x, xh, M * KC);
    // k2: dequant + LoRA fold -> W_eff f16
    dequant_fold<<<dim3(KC / 256, NC), 256, 0, stream>>>(codes, absmax, lora_A, lora_B, wh);
    // k3: WMMA GEMM
    qlora_gemm<<<dim3(NC / 128, M / 128), 256, 0, stream>>>(xh, wh, out);
}
